// UNet_1967095021868
// MI455X (gfx1250) — matrix operand order repaired and checked by reading, not compiled, not run on hardware
//
#include <hip/hip_runtime.h>
#include <stdint.h>
#include <stddef.h>

// ---------------- CDNA5 WMMA types ----------------
typedef __attribute__((ext_vector_type(16))) __bf16 v16bf;
typedef __attribute__((ext_vector_type(8)))  float  v8f;
// ---------------- TDM descriptor groups (probe-verified builtin signatures) ----------------
typedef __attribute__((ext_vector_type(4))) unsigned tdm_v4u;
typedef __attribute__((ext_vector_type(8))) int      tdm_v8i;
typedef __attribute__((ext_vector_type(4))) int      tdm_v4i;

#define BIGF 1000000000.0f

__device__ __forceinline__ __bf16 f2bf(float f) {
  union { float f; uint32_t u; } a; a.f = f;
  uint32_t u = a.u + 0x7FFFu + ((a.u >> 16) & 1u);   // round-to-nearest-even
  union { uint16_t s; __bf16 b; } c; c.s = (uint16_t)(u >> 16);
  return c.b;
}

// Issue one TDM 2D tile load: 16 rows x 32 fp32 from global (row stride = lda elements)
// into LDS at byte offset ldsOff, with LDS padding of 1 dword every 32 dwords
// (=> 33-float row stride in LDS, bank-conflict-free column reads).
// OOB rows/cols (beyond remM/remK) are zero-filled by the TDM.
__device__ __forceinline__ void tdm_load_a_tile(const float* ga, unsigned ldsOff,
                                                unsigned remK, unsigned remM,
                                                unsigned long long stride0) {
  unsigned long long g = (unsigned long long)(uintptr_t)ga;
  tdm_v4u g0;
  g0[0] = 1u;                                            // count=1 (valid user D#)
  g0[1] = ldsOff;                                        // lds_addr (bytes)
  g0[2] = (unsigned)(g & 0xFFFFFFFFull);                 // global_addr[31:0]
  g0[3] = (unsigned)((g >> 32) & 0x1FFFFFFull) | (2u << 30); // global_addr[56:32] | type=2
  tdm_v8i g1;
  g1[0] = (int)(0x20000u | (1u << 20) | (4u << 22));     // data_size=4B | pad_enable | pad_interval=32dw (pad_amount=1dw)
  g1[1] = (int)((remK & 0xFFFFu) << 16);                 // tensor_dim0[15:0] @ bits63:48
  g1[2] = (int)(((remK >> 16) & 0xFFFFu) | ((remM & 0xFFFFu) << 16)); // dim0 hi | dim1 lo
  g1[3] = (int)(((remM >> 16) & 0xFFFFu) | (32u << 16)); // dim1 hi | tile_dim0=32
  g1[4] = 16;                                            // tile_dim1=16 rows (tile_dim2=0)
  g1[5] = (int)(unsigned)(stride0 & 0xFFFFFFFFull);      // tensor_dim0_stride lo
  g1[6] = (int)(unsigned)((stride0 >> 32) & 0xFFFFull);  // stride hi (dim1_stride=0)
  g1[7] = 0;
  tdm_v4i gz = {0, 0, 0, 0};                             // groups 2/3 unused (2D tensor)
#if __clang_major__ >= 23
  tdm_v8i g4 = {0, 0, 0, 0, 0, 0, 0, 0};
  __builtin_amdgcn_tensor_load_to_lds(g0, g1, gz, gz, g4, 0);
#else
  __builtin_amdgcn_tensor_load_to_lds(g0, g1, gz, gz, 0);
#endif
}

// =====================================================================
// WMMA GEMM: C[M,N] = act( A[M,K] @ W[N,K]^T + bias[col % bias_mod] ) (+= C if accum)
// One wave per 16x16 C tile. A tiles staged to LDS by the Tensor Data Mover
// (double-buffered, s_wait_tensorcnt pipelined); B read as float4 vectors.
// bf16 operands, f32 accumulation via V_WMMA_F32_16X16X32_BF16.
// =====================================================================
#define LDSROW 33      // 32 floats + 1 pad dword inserted by TDM
#define ABUF   544     // 16*33 = 528, rounded up
__global__ void k_gemm(const float* __restrict__ A, long long lda, long long sA,
                       const float* __restrict__ W, long long ldw,
                       const float* __restrict__ bias, int bias_mod,
                       float* __restrict__ C, long long ldc, long long sC,
                       int M, int N, int K, int act, int accum)
{
  __shared__ float ldsA[8 * 2 * ABUF];   // 8 waves x 2 buffers, ~34KB of the WGP's 320KB
  int b = blockIdx.y;
  A += (size_t)b * sA;
  C += (size_t)b * sC;
  int tilesN = (N + 15) >> 4;
  int tilesM = (M + 15) >> 4;
  int w = threadIdx.x >> 5;
  int tile = blockIdx.x * (blockDim.x >> 5) + w;
  if (tile >= tilesM * tilesN) return;
  int m0 = (tile / tilesN) << 4;
  int n0 = (tile % tilesN) << 4;
  int lane = threadIdx.x & 31;
  int lo = lane & 15;
  int hi = lane >> 4;
  int bcol = n0 + lo;                      // B: one output column per lane
  float* slice = &ldsA[w * 2 * ABUF];

  int T = (K + 31) >> 5;
  const float* Abase = A + (size_t)m0 * lda;
  // prologue: TDM-stage tile 0
  tdm_load_a_tile(Abase, (unsigned)(size_t)&slice[0], (unsigned)K, (unsigned)(M - m0),
                  (unsigned long long)lda);
  v8f acc = {0.f, 0.f, 0.f, 0.f, 0.f, 0.f, 0.f, 0.f};
  for (int i = 0; i < T; ++i) {
    int kk = i << 5;
    if (i + 1 < T) {   // issue next tile into the other buffer, then wait for tile i
      tdm_load_a_tile(Abase + (kk + 32), (unsigned)(size_t)&slice[((i + 1) & 1) * ABUF],
                      (unsigned)(K - (kk + 32)), (unsigned)(M - m0), (unsigned long long)lda);
      __builtin_amdgcn_s_wait_tensorcnt(1);
    } else {
      __builtin_amdgcn_s_wait_tensorcnt(0);
    }
    const float* at = &slice[(i & 1) * ABUF];
    v16bf af, bfv;
    // A fragment from padded LDS tile (TDM already zero-filled any OOB rows/cols)
    #pragma unroll
    for (int j = 0; j < 16; ++j) {
      int kl = hi * 16 + j;  // contiguous k order (k = 16*hi + j), matches B fragment
      af[j] = f2bf(at[lo * LDSROW + kl]);
    }
    // B fragment: lane's column is contiguous in W -> 4x float4 on the fast path
    if (bcol < N && kk + 32 <= K) {
      const float4* Wv = (const float4*)(W + (size_t)bcol * ldw + kk + hi * 16);
      float4 w0 = Wv[0], w1 = Wv[1], w2 = Wv[2], w3 = Wv[3];
      bfv[0]  = f2bf(w0.x); bfv[1]  = f2bf(w0.y); bfv[2]  = f2bf(w0.z); bfv[3]  = f2bf(w0.w);
      bfv[4]  = f2bf(w1.x); bfv[5]  = f2bf(w1.y); bfv[6]  = f2bf(w1.z); bfv[7]  = f2bf(w1.w);
      bfv[8]  = f2bf(w2.x); bfv[9]  = f2bf(w2.y); bfv[10] = f2bf(w2.z); bfv[11] = f2bf(w2.w);
      bfv[12] = f2bf(w3.x); bfv[13] = f2bf(w3.y); bfv[14] = f2bf(w3.z); bfv[15] = f2bf(w3.w);
    } else {
      #pragma unroll
      for (int j = 0; j < 16; ++j) {
        int kb = kk + hi * 16 + j;
        float bv = (bcol < N && kb < K) ? W[(size_t)bcol * ldw + kb] : 0.f;
        bfv[j] = f2bf(bv);
      }
    }
    acc = __builtin_amdgcn_wmma_f32_16x16x32_bf16(false, af, false, bfv,
                                                  (short)0, acc, false, false);
  }
  #pragma unroll
  for (int r = 0; r < 8; ++r) {
    int row = m0 + hi * 8 + r;   // C layout: VGPR r -> M=r (lanes0-15), M=8+r (lanes16-31)
    int col = n0 + lo;
    if (row < M && col < N) {
      float v = acc[r];
      if (bias) v += bias[bias_mod ? (col % bias_mod) : col];
      if (act == 1) v = v > 0.f ? v : 0.f;
      else if (act == 2) v = v > 0.f ? v : 0.2f * v;
      size_t o = (size_t)row * ldc + col;
      if (accum) v += C[o];
      C[o] = v;
    }
  }
}

// ---------------- layout transposes ----------------
__global__ void k_cn_to_nc(const float* __restrict__ in, float* __restrict__ out,
                           int C, int N, long long total) {
  long long i = (long long)blockIdx.x * blockDim.x + threadIdx.x;
  if (i >= total) return;
  int b = (int)(i / ((long long)C * N));
  long long rem = i % ((long long)C * N);
  int c = (int)(rem / N), n = (int)(rem % N);
  out[((size_t)b * N + n) * C + c] = in[i];
}
__global__ void k_nc_to_cn(const float* __restrict__ in, float* __restrict__ out,
                           int N, int C, long long total) {
  long long i = (long long)blockIdx.x * blockDim.x + threadIdx.x;
  if (i >= total) return;
  int b = (int)(i / ((long long)N * C));
  long long rem = i % ((long long)N * C);
  int n = (int)(rem / C), c = (int)(rem % C);
  out[((size_t)b * C + c) * N + n] = in[i];
}
__global__ void k_copy(float* __restrict__ dst, const float* __restrict__ src, long long n) {
  long long i = (long long)blockIdx.x * blockDim.x + threadIdx.x;
  if (i < n) dst[i] = src[i];
}
__global__ void k_copy2(const float* __restrict__ src, long long sS,
                        float* __restrict__ dst, long long sD, int n, int B) {
  long long t = (long long)blockIdx.x * blockDim.x + threadIdx.x;
  if (t >= (long long)n * B) return;
  int b = (int)(t / n), i = (int)(t % n);
  dst[(size_t)b * sD + i] = src[(size_t)b * sS + i];
}

// ---------------- FPS: LDS-resident distances + argmax tree ----------------
__global__ void k_fps(const float* __restrict__ ptsT, int N, int m, int* __restrict__ idx) {
  __shared__ float dist[2048];
  __shared__ float rv[256];
  __shared__ int   ri[256];
  int b = blockIdx.x, tid = threadIdx.x;
  const float* P = ptsT + (size_t)b * N * 3;
  if (tid == 0) idx[(size_t)b * m] = 0;
  float x0 = P[0], y0 = P[1], z0 = P[2];
  for (int i = tid; i < N; i += 256) {
    float dx = P[i*3] - x0, dy = P[i*3+1] - y0, dz = P[i*3+2] - z0;
    dist[i] = dx*dx + dy*dy + dz*dz;
  }
  for (int s = 1; s < m; ++s) {
    __syncthreads();
    float best = -1.f; int bi = 0;
    for (int i = tid; i < N; i += 256) {
      float d = dist[i];
      if (d > best) { best = d; bi = i; }
    }
    rv[tid] = best; ri[tid] = bi;
    __syncthreads();
    for (int offr = 128; offr > 0; offr >>= 1) {
      if (tid < offr && rv[tid + offr] > rv[tid]) { rv[tid] = rv[tid + offr]; ri[tid] = ri[tid + offr]; }
      __syncthreads();
    }
    int j = ri[0];
    if (tid == 0) idx[(size_t)b * m + s] = j;
    float xj = P[j*3], yj = P[j*3+1], zj = P[j*3+2];
    for (int i = tid; i < N; i += 256) {
      float dx = P[i*3] - xj, dy = P[i*3+1] - yj, dz = P[i*3+2] - zj;
      float d = dx*dx + dy*dy + dz*dz;
      if (d < dist[i]) dist[i] = d;
    }
  }
}

// ---------------- brute-force kNN (k=16), register insertion sort ----------------
__global__ void k_knn(const float* __restrict__ q, const float* __restrict__ r,
                      int M, int N, int* __restrict__ nbr, int B) {
  int t = blockIdx.x * blockDim.x + threadIdx.x;
  if (t >= B * M) return;
  int b = t / M, m = t % M;
  const float* Q = q + ((size_t)b * M + m) * 3;
  const float* R = r + (size_t)b * N * 3;
  float qx = Q[0], qy = Q[1], qz = Q[2];
  float bd[16]; int bi[16];
  #pragma unroll
  for (int s = 0; s < 16; ++s) { bd[s] = 3.4e38f; bi[s] = 0; }
  for (int j = 0; j < N; ++j) {
    float dx = R[j*3] - qx, dy = R[j*3+1] - qy, dz = R[j*3+2] - qz;
    float d = dx*dx + dy*dy + dz*dz;
    if (d < bd[15]) {
      bd[15] = d; bi[15] = j;
      #pragma unroll
      for (int s = 15; s > 0; --s) {
        if (bd[s] < bd[s-1]) {
          float tv = bd[s]; bd[s] = bd[s-1]; bd[s-1] = tv;
          int ti = bi[s]; bi[s] = bi[s-1]; bi[s-1] = ti;
        }
      }
    }
  }
  int* O = nbr + ((size_t)b * M + m) * 16;
  #pragma unroll
  for (int s = 0; s < 16; ++s) O[s] = bi[s];
}

// ---------------- nearest anchor (argmin over 128) ----------------
__global__ void k_nearest(const float* __restrict__ ptsT, const float* __restrict__ anchT,
                          int N, int* __restrict__ out, int B) {
  int t = blockIdx.x * blockDim.x + threadIdx.x;
  if (t >= B * N) return;
  int b = t / N, n = t % N;
  const float* P = ptsT + ((size_t)b * N + n) * 3;
  const float* A = anchT + (size_t)b * 128 * 3;
  float px = P[0], py = P[1], pz = P[2];
  float best = 3.4e38f; int bi = 0;
  for (int j = 0; j < 128; ++j) {
    float dx = A[j*3] - px, dy = A[j*3+1] - py, dz = A[j*3+2] - pz;
    float d = dx*dx + dy*dy + dz*dz;
    if (d < best) { best = d; bi = j; }
  }
  out[t] = bi;
}

// ---------------- geodesic: 128x128 matrix resident in LDS, Floyd–Warshall ----------------
__global__ void k_geodesic(const float* __restrict__ anch, float* __restrict__ out) {
  __shared__ float D[128 * 128];  // 64KB of the WGP's 320KB LDS
  int b = blockIdx.x, tid = threadIdx.x;
  const float* A = anch + (size_t)b * 128 * 3;
  for (int i = tid; i < 16384; i += 256) {
    int r = i >> 7, c = i & 127;
    float dx = A[r*3] - A[c*3], dy = A[r*3+1] - A[c*3+1], dz = A[r*3+2] - A[c*3+2];
    float s = dx*dx + dy*dy + dz*dz;
    D[i] = sqrtf(s > 0.f ? s : 0.f);
  }
  __syncthreads();
  float v0 = BIGF, v1 = BIGF, v2 = BIGF; int i0 = 0, i1 = 0, i2 = 0;
  if (tid < 128) {
    int r = tid;
    for (int c = 0; c < 128; ++c) {
      float d = D[r*128 + c] + (c == r ? BIGF : 0.f);
      if (d < v0)      { v2 = v1; i2 = i1; v1 = v0; i1 = i0; v0 = d; i0 = c; }
      else if (d < v1) { v2 = v1; i2 = i1; v1 = d; i1 = c; }
      else if (d < v2) { v2 = d; i2 = c; }
    }
  }
  __syncthreads();
  for (int i = tid; i < 16384; i += 256) D[i] = BIGF;
  __syncthreads();
  if (tid < 128) {
    D[tid*128 + i0] = v0; D[tid*128 + i1] = v1; D[tid*128 + i2] = v2;
  }
  __syncthreads();
  for (int i = tid; i < 16384; i += 256) {      // W = min(W, W^T); diag = 0
    int r = i >> 7, c = i & 127;
    if (r < c) {
      float mnv = fminf(D[r*128 + c], D[c*128 + r]);
      D[r*128 + c] = mnv; D[c*128 + r] = mnv;
    } else if (r == c) D[i] = 0.f;
  }
  __syncthreads();
  for (int p = 0; p < 128; ++p) {               // Floyd–Warshall, all in LDS
    for (int i = tid; i < 16384; i += 256) {
      int r = i >> 7, c = i & 127;
      float nd = D[r*128 + p] + D[p*128 + c];
      if (nd < D[i]) D[i] = nd;
    }
    __syncthreads();
  }
  float* O = out + (size_t)b * 16384;
  for (int i = tid; i < 16384; i += 256) {
    float d = D[i];
    O[i] = (d > BIGF * 0.5f) ? 0.f : d;
  }
}

// ---------------- gathers ----------------
__global__ void k_gather_rows(const float* __restrict__ in, const int* __restrict__ idx,
                              float* __restrict__ out, int N, int M, int C, long long total) {
  long long t = (long long)blockIdx.x * blockDim.x + threadIdx.x;
  if (t >= total) return;
  int c = (int)(t % C);
  long long r = t / C;
  int m = (int)(r % M), b = (int)(r / M);
  out[t] = in[((size_t)b * N + idx[(size_t)b * M + m]) * C + c];
}
__global__ void k_gather_int(const int* __restrict__ src, const int* __restrict__ idx,
                             int* __restrict__ out, int N, int M, long long total) {
  long long t = (long long)blockIdx.x * blockDim.x + threadIdx.x;
  if (t >= total) return;
  int b = (int)(t / M);
  out[t] = src[(size_t)b * N + idx[t]];
}
__global__ void k_gather_nbr(const float* __restrict__ in, const int* __restrict__ idx,
                             float* __restrict__ out, int N, int C) {
  long long t = (long long)blockIdx.x * blockDim.x + threadIdx.x;
  if (t >= (long long)N * 16 * C) return;
  int c = (int)(t % C);
  long long r = t / C;
  int k = (int)(r % 16), n = (int)(r / 16);
  out[t] = in[(size_t)idx[n*16 + k] * C + c];
}

// ---------------- DGCNN edge features ----------------
__global__ void k_edge(const float* __restrict__ feaT, const int* __restrict__ ctrIdx,
                       const int* __restrict__ nbr, float* __restrict__ edge,
                       int N, int M, int Kn, int C, long long total) {
  long long t = (long long)blockIdx.x * blockDim.x + threadIdx.x;
  if (t >= total) return;
  int c = (int)(t % C);
  long long r = t / C;
  int kk = (int)(r % Kn);
  long long r2 = r / Kn;
  int m = (int)(r2 % M), b = (int)(r2 / M);
  int ci = ctrIdx[(size_t)b * M + m];
  int nj = nbr ? nbr[((size_t)b * M + m) * Kn + kk] : kk;
  const float* fb = feaT + (size_t)b * N * C;
  float ctr = fb[(size_t)ci * C + c];
  float g   = fb[(size_t)nj * C + c];
  float* E = edge + (((size_t)b * M + m) * Kn + kk) * (2 * C);
  E[c] = g - ctr;
  E[C + c] = ctr;
}
__global__ void k_maxk(const float* __restrict__ h, float* __restrict__ out,
                       int M, int Kn, int C, long long total) {
  long long t = (long long)blockIdx.x * blockDim.x + threadIdx.x;
  if (t >= total) return;
  int c = (int)(t % C);
  long long r = t / C;
  int m = (int)(r % M), b = (int)(r / M);
  const float* hp = h + (((size_t)b * M + m) * Kn) * C + c;
  float mx = -3.4e38f;
  for (int k = 0; k < Kn; ++k) mx = fmaxf(mx, hp[(size_t)k * C]);
  out[t] = mx;
}

// ---------------- transformer elementwise pieces (single batch, d=64, k=16) ----------------
__global__ void k_posrel(const float* __restrict__ pos, const int* __restrict__ idx,
                         float* __restrict__ pr, int N) {
  long long t = (long long)blockIdx.x * blockDim.x + threadIdx.x;
  if (t >= (long long)N * 48) return;
  int c = (int)(t % 3);
  long long r = t / 3;
  int k = (int)(r % 16), n = (int)(r / 16);
  pr[t] = pos[(size_t)n*3 + c] - pos[(size_t)idx[n*16 + k]*3 + c];
}
__global__ void k_qkpe(const float* __restrict__ q, const float* __restrict__ kg,
                       const float* __restrict__ pe, float* __restrict__ qk, int N) {
  long long t = (long long)blockIdx.x * blockDim.x + threadIdx.x;
  if (t >= (long long)N * 16 * 64) return;
  int d = (int)(t % 64);
  int n = (int)(t / (16 * 64));
  qk[t] = q[(size_t)n*64 + d] - kg[t] + pe[t];
}
__global__ void k_attn(const float* __restrict__ a, const float* __restrict__ vg,
                       const float* __restrict__ pe, float* __restrict__ agg, int N) {
  int t = blockIdx.x * blockDim.x + threadIdx.x;
  if (t >= N * 64) return;
  int n = t / 64, d = t % 64;
  const float* ap = a  + (size_t)n * 16 * 64 + d;
  const float* vp = vg + (size_t)n * 16 * 64 + d;
  const float* pp = pe + (size_t)n * 16 * 64 + d;
  float mx = -3.4e38f;
  #pragma unroll
  for (int k = 0; k < 16; ++k) mx = fmaxf(mx, ap[k * 64]);
  float s = 0.f;
  #pragma unroll
  for (int k = 0; k < 16; ++k) s += __expf(ap[k * 64] - mx);
  float inv = 1.f / s;
  float accv = 0.f;
  #pragma unroll
  for (int k = 0; k < 16; ++k)
    accv += __expf(ap[k * 64] - mx) * inv * (vp[k * 64] + pp[k * 64]);
  agg[t] = accv;
}

// ---------------- decoder helpers ----------------
__global__ void k_reordw(const float* __restrict__ W, float* __restrict__ Wr,
                         int C, int O, int Kk, long long total) {
  // W (C,O,Kk) -> Wr (Kk,O,C) so conv_transpose == row-major GEMM
  long long i = (long long)blockIdx.x * blockDim.x + threadIdx.x;
  if (i >= total) return;
  int kk = (int)(i % Kk);
  long long r = i / Kk;
  int o = (int)(r % O), c = (int)(r / O);
  Wr[((size_t)kk * O + o) * C + c] = W[i];
}
__global__ void k_u0cat(const float* __restrict__ u0T, const float* __restrict__ l4T,
                        float* __restrict__ out, long long total) {
  long long i = (long long)blockIdx.x * blockDim.x + threadIdx.x;
  if (i >= total) return;
  int c = (int)(i % 640);
  long long r = i / 640;
  int tk = (int)(r % 64), b = (int)(r / 64);
  out[i] = (c < 256) ? u0T[((size_t)b * 64 + tk) * 256 + c] : l4T[(size_t)b * 384 + (c - 256)];
}

// =====================================================================
// host-side orchestration
// =====================================================================
static inline dim3 blks(long long n) { return dim3((unsigned)((n + 255) / 256)); }

static void gemm(hipStream_t st,
                 const float* A, long long lda, long long sA,
                 const float* W, long long ldw,
                 const float* bias, int bias_mod,
                 float* C, long long ldc, long long sC,
                 int M, int N, int K, int B, int act, int accum) {
  long long tiles = (long long)((M + 15) / 16) * ((N + 15) / 16);
  dim3 g((unsigned)((tiles + 7) / 8), (unsigned)B);
  k_gemm<<<g, 256, 0, st>>>(A, lda, sA, W, ldw, bias, bias_mod, C, ldc, sC, M, N, K, act, accum);
}

struct TrScratch {
  float *x, *q, *k, *v, *kg, *vg, *pe, *peh, *qk, *amid, *a, *agg, *pr;
  int* idx;
};

// Point transformer block, in-place on feaT (B,N,C); batch-sequential scratch reuse.
static void run_transformer(hipStream_t st, float* feaT, const float* posT,
                            int Bn, int N, int C, const float* const* P, const TrScratch& S) {
  for (int b = 0; b < Bn; ++b) {
    const float* pos = posT + (size_t)b * N * 3;
    float* fea = feaT + (size_t)b * N * C;
    gemm(st, fea,  C,  0, P[0],  C,  P[1],  0, S.x, 64, 0, N, 64, C,  1, 0, 0);   // x = Ws fea + bs
    gemm(st, S.x,  64, 0, P[2],  64, P[3],  0, S.q, 64, 0, N, 64, 64, 1, 0, 0);   // q
    gemm(st, S.x,  64, 0, P[4],  64, P[5],  0, S.k, 64, 0, N, 64, 64, 1, 0, 0);   // k
    gemm(st, S.x,  64, 0, P[6],  64, P[7],  0, S.v, 64, 0, N, 64, 64, 1, 0, 0);   // v
    k_knn<<<blks(N), 256, 0, st>>>(pos, pos, N, N, S.idx, 1);
    k_gather_nbr<<<blks((long long)N*16*64), 256, 0, st>>>(S.k, S.idx, S.kg, N, 64);
    k_gather_nbr<<<blks((long long)N*16*64), 256, 0, st>>>(S.v, S.idx, S.vg, N, 64);
    k_posrel<<<blks((long long)N*48), 256, 0, st>>>(pos, S.idx, S.pr, N);
    gemm(st, S.pr,   3,   0, P[8],  3,   P[9],  0, S.peh,  64,  0, N*16, 64,  3,   1, 1, 0);
    gemm(st, S.peh,  64,  0, P[10], 64,  P[11], 0, S.pe,   64,  0, N*16, 64,  64,  1, 0, 0);
    k_qkpe<<<blks((long long)N*16*64), 256, 0, st>>>(S.q, S.kg, S.pe, S.qk, N);
    gemm(st, S.qk,   64,  0, P[12], 64,  P[13], 0, S.amid, 256, 0, N*16, 256, 64,  1, 1, 0);
    gemm(st, S.amid, 256, 0, P[14], 256, P[15], 0, S.a,    64,  0, N*16, 64,  256, 1, 0, 0);
    k_attn<<<blks((long long)N*64), 256, 0, st>>>(S.a, S.vg, S.pe, S.agg, N);
    gemm(st, S.agg,  64,  0, P[16], 64,  P[17], 0, fea,    C,   0, N,    C,   64,  1, 0, 1); // + identity
  }
}

extern "C" void kernel_launch(void* const* d_in, const int* in_sizes, int n_in,
                              void* d_out, int out_size, void* d_ws, size_t ws_size,
                              hipStream_t stream) {
  (void)in_sizes; (void)n_in; (void)out_size; (void)ws_size;
  auto F = [&](int i) { return (const float*)d_in[i]; };
  const float* PC = F(0);
  const float *sa1_W1 = F(1),  *sa1_b1 = F(2),  *sa1_W2 = F(3),  *sa1_b2 = F(4);
  const float *sa2_W1 = F(5),  *sa2_b1 = F(6),  *sa2_W2 = F(7),  *sa2_b2 = F(8);
  const float *sa4_W1 = F(9),  *sa4_b1 = F(10), *sa4_W2 = F(11), *sa4_b2 = F(12);
  const float* TR1[18]; for (int t = 0; t < 18; ++t) TR1[t] = F(13 + t);
  const float* TR2[18]; for (int t = 0; t < 18; ++t) TR2[t] = F(31 + t);
  const float* TR4[18]; for (int t = 0; t < 18; ++t) TR4[t] = F(49 + t);
  const float *p1W1 = F(67), *p1b1 = F(68), *p1W2 = F(69), *p1b2 = F(70);
  const float *p2W1 = F(71), *p2b1 = F(72), *p2W2 = F(73), *p2b2 = F(74);
  const float *ps0W = F(75), *ps0b = F(76), *ps1W = F(77), *ps1b = F(78), *ps2W = F(79), *ps2b = F(80);
  const float *m1W1 = F(81), *m1b1 = F(82), *m1W2 = F(83), *m1b2 = F(84), *m1Ws = F(85), *m1bs = F(86);
  const float *m3W1 = F(87), *m3b1 = F(88), *m3W2 = F(89), *m3b2 = F(90), *m3Ws = F(91), *m3bs = F(92);
  float* out = (float*)d_out;
  hipStream_t st = stream;
  const int B = 16, N0 = 2048;

  // deterministic bump allocator over d_ws
  char* base = (char*)d_ws;
  size_t off = 0;
  auto alloc  = [&](size_t bytes) -> void* { void* p = base + off; off = (off + bytes + 255) & ~(size_t)255; return p; };
  auto allocf = [&](size_t n) { return (float*)alloc(n * 4); };
  auto alloci = [&](size_t n) { return (int*)alloc(n * 4); };

  float* ptsT     = allocf((size_t)B * N0 * 3);
  int*   aidx     = alloci((size_t)B * 128);
  float* anchT    = allocf((size_t)B * 128 * 3);
  int*   mapIdx   = alloci((size_t)B * N0);
  float* anchDist = allocf((size_t)B * 128 * 128);
  int*   fi1      = alloci((size_t)B * 512);
  float* l1xyzT   = allocf((size_t)B * 512 * 3);
  int*   nbr1     = alloci((size_t)B * 512 * 16);
  float* edge1    = allocf((size_t)B * 512 * 16 * 6);
  float* h1       = allocf((size_t)B * 512 * 16 * 8);
  float* hm1      = allocf((size_t)B * 512 * 8);
  float* l1feaT   = allocf((size_t)B * 512 * 128);
  int*   mi1      = alloci((size_t)B * 512);
  float* emb      = allocf((size_t)B * 512 * 128);
  float* embh     = allocf((size_t)B * 512 * 256);
  int*   fi2      = alloci((size_t)B * 128);
  float* l2xyzT   = allocf((size_t)B * 128 * 3);
  int*   nbr2     = alloci((size_t)B * 128 * 16);
  float* edge2    = allocf((size_t)B * 128 * 16 * 256);
  float* h2       = allocf((size_t)B * 128 * 16 * 128);
  float* hm2      = allocf((size_t)B * 128 * 128);
  float* l2feaT   = allocf((size_t)B * 128 * 256);
  int*   mi2      = alloci((size_t)B * 128);
  int*   fi3      = alloci((size_t)B);
  float* edge4    = allocf((size_t)B * 128 * 512);
  float* h4       = allocf((size_t)B * 128 * 256);
  float* hm4      = allocf((size_t)B * 256);
  float* l4T      = allocf((size_t)B * 384);
  float* Wr0      = allocf((size_t)16384 * 384);
  float* Wr1      = allocf((size_t)2 * 256 * 256);
  float* Wr2      = allocf((size_t)2 * 128 * 256);
  float* u0T      = allocf((size_t)B * 64 * 256);
  float* u0cat    = allocf((size_t)B * 64 * 640);
  float* mh       = allocf((size_t)B * 64 * 256);
  float* u0rT     = allocf((size_t)B * 64 * 256);
  float* u1catT   = allocf((size_t)B * 256 * 256);
  float* u2T      = allocf((size_t)B * 512 * 128);
  float* u2xyzT   = allocf((size_t)B * 1024 * 3);
  float* m3h      = allocf((size_t)B * 512 * 32);
  float* u2feaT   = allocf((size_t)B * 1024 * 128);
  TrScratch S;
  S.x    = allocf((size_t)1024 * 64);
  S.q    = allocf((size_t)1024 * 64);
  S.k    = allocf((size_t)1024 * 64);
  S.v    = allocf((size_t)1024 * 64);
  S.idx  = alloci((size_t)1024 * 16);
  S.kg   = allocf((size_t)1024 * 16 * 64);
  S.vg   = allocf((size_t)1024 * 16 * 64);
  S.pe   = allocf((size_t)1024 * 16 * 64);
  S.peh  = allocf((size_t)1024 * 16 * 64);
  S.qk   = allocf((size_t)1024 * 16 * 64);
  S.amid = allocf((size_t)1024 * 16 * 256);
  S.a    = allocf((size_t)1024 * 16 * 64);
  S.agg  = allocf((size_t)1024 * 64);
  S.pr   = allocf((size_t)1024 * 16 * 3);

  // ---- anchors / geodesic embedding ----
  k_cn_to_nc<<<blks((long long)B*3*N0), 256, 0, st>>>(PC, ptsT, 3, N0, (long long)B*3*N0);
  k_fps<<<B, 256, 0, st>>>(ptsT, N0, 128, aidx);
  k_gather_rows<<<blks((long long)B*128*3), 256, 0, st>>>(ptsT, aidx, anchT, N0, 128, 3, (long long)B*128*3);
  k_nearest<<<blks((long long)B*N0), 256, 0, st>>>(ptsT, anchT, N0, mapIdx, B);
  k_geodesic<<<B, 256, 0, st>>>(anchT, anchDist);

  // ---- sa1: dgcnn_grouper(pc, pc, 512, 16) ----
  k_fps<<<B, 256, 0, st>>>(ptsT, N0, 512, fi1);
  k_gather_rows<<<blks((long long)B*512*3), 256, 0, st>>>(ptsT, fi1, l1xyzT, N0, 512, 3, (long long)B*512*3);
  k_knn<<<blks((long long)B*512), 256, 0, st>>>(l1xyzT, ptsT, 512, N0, nbr1, B);
  k_edge<<<blks((long long)B*512*16*3), 256, 0, st>>>(ptsT, fi1, nbr1, edge1, N0, 512, 16, 3, (long long)B*512*16*3);
  gemm(st, edge1, 6, 512*16*6, sa1_W1, 6, sa1_b1, 0, h1, 8, 512*16*8, 512*16, 8, 6, B, 2, 0);
  k_maxk<<<blks((long long)B*512*8), 256, 0, st>>>(h1, hm1, 512, 16, 8, (long long)B*512*8);
  gemm(st, hm1, 8, 512*8, sa1_W2, 8, sa1_b2, 0, l1feaT, 128, 512*128, 512, 128, 8, B, 2, 0);

  // ---- geodesic embedding proj1, added to l1_fea ----
  k_gather_int<<<blks((long long)B*512), 256, 0, st>>>(mapIdx, fi1, mi1, N0, 512, (long long)B*512);
  k_gather_rows<<<blks((long long)B*512*128), 256, 0, st>>>(anchDist, mi1, emb, 128, 512, 128, (long long)B*512*128);
  gemm(st, emb,  128, 512*128, p1W1, 128, p1b1, 0, embh,   256, 512*256, 512, 256, 128, B, 1, 0);
  gemm(st, embh, 256, 512*256, p1W2, 256, p1b2, 0, l1feaT, 128, 512*128, 512, 128, 256, B, 0, 1);

  run_transformer(st, l1feaT, l1xyzT, B, 512, 128, TR1, S);

  // ---- sa2: dgcnn_grouper(l1, 128, 16) ----
  k_fps<<<B, 256, 0, st>>>(l1xyzT, 512, 128, fi2);
  k_gather_rows<<<blks((long long)B*128*3), 256, 0, st>>>(l1xyzT, fi2, l2xyzT, 512, 128, 3, (long long)B*128*3);
  k_knn<<<blks((long long)B*128), 256, 0, st>>>(l2xyzT, l1xyzT, 128, 512, nbr2, B);
  k_edge<<<blks((long long)B*128*16*128), 256, 0, st>>>(l1feaT, fi2, nbr2, edge2, 512, 128, 16, 128, (long long)B*128*16*128);
  gemm(st, edge2, 256, 128*16*256, sa2_W1, 256, sa2_b1, 0, h2, 128, 128*16*128, 128*16, 128, 256, B, 2, 0);
  k_maxk<<<blks((long long)B*128*128), 256, 0, st>>>(h2, hm2, 128, 16, 128, (long long)B*128*128);
  gemm(st, hm2, 128, 128*128, sa2_W2, 128, sa2_b2, 0, l2feaT, 256, 128*256, 128, 256, 128, B, 2, 0);

  k_gather_int<<<blks((long long)B*128), 256, 0, st>>>(mi1, fi2, mi2, 512, 128, (long long)B*128);
  k_gather_rows<<<blks((long long)B*128*128), 256, 0, st>>>(anchDist, mi2, emb, 128, 128, 128, (long long)B*128*128);
  gemm(st, emb,  128, 128*128, p2W1, 128, p2b1, 0, embh,   256, 128*256, 128, 256, 128, B, 1, 0);
  gemm(st, embh, 256, 128*256, p2W2, 256, p2b2, 0, l2feaT, 256, 128*256, 128, 256, 256, B, 0, 1);

  run_transformer(st, l2feaT, l2xyzT, B, 128, 256, TR2, S);

  // ---- sa4: dgcnn_grouper(l2, m=1, k=128) — neighbor set = all points, order-free under max ----
  k_fps<<<B, 256, 0, st>>>(l2xyzT, 128, 1, fi3);
  k_edge<<<blks((long long)B*128*256), 256, 0, st>>>(l2feaT, fi3, (const int*)nullptr, edge4, 128, 1, 128, 256, (long long)B*128*256);
  gemm(st, edge4, 512, 128*512, sa4_W1, 512, sa4_b1, 0, h4, 256, 128*256, 128, 256, 512, B, 2, 0);
  k_maxk<<<blks((long long)B*256), 256, 0, st>>>(h4, hm4, 1, 128, 256, (long long)B*256);
  gemm(st, hm4, 256, 256, sa4_W2, 256, sa4_b2, 0, l4T, 384, 384, 1, 384, 256, B, 2, 0);

  // ---- decoder ----
  k_reordw<<<blks((long long)384*256*64), 256, 0, st>>>(ps0W, Wr0, 384, 256, 64, (long long)384*256*64);
  gemm(st, l4T, 384, 0, Wr0, 384, ps0b, 256, u0T, 16384, 0, B, 16384, 384, 1, 0, 0);
  k_u0cat<<<blks((long long)B*64*640), 256, 0, st>>>(u0T, l4T, u0cat, (long long)B*64*640);
  gemm(st, u0cat, 640, 64*640, m1Ws, 640, m1bs, 0, u0rT, 256, 64*256, 64, 256, 640, B, 0, 0);
  gemm(st, u0cat, 640, 64*640, m1W1, 640, m1b1, 0, mh,   256, 64*256, 64, 256, 640, B, 1, 0);
  gemm(st, mh,    256, 64*256, m1W2, 256, m1b2, 0, u0rT, 256, 64*256, 64, 256, 256, B, 0, 1);
  k_reordw<<<blks((long long)256*256*2), 256, 0, st>>>(ps1W, Wr1, 256, 256, 2, (long long)256*256*2);
  k_copy2<<<blks((long long)B*128*256), 256, 0, st>>>(l2feaT, 128*256, u1catT, 256*256, 128*256, B);
  for (int j = 0; j < 2; ++j)
    gemm(st, u0rT, 256, 64*256, Wr1 + (size_t)j*256*256, 256, ps1b, 0,
         u1catT + 128*256 + j*256, 512, 256*256, 64, 256, 256, B, 0, 0);
  k_reordw<<<blks((long long)256*128*2), 256, 0, st>>>(ps2W, Wr2, 256, 128, 2, (long long)256*128*2);
  for (int j = 0; j < 2; ++j)
    gemm(st, u1catT, 256, 256*256, Wr2 + (size_t)j*128*256, 256, ps2b, 0,
         u2T + j*128, 256, 512*128, 256, 128, 256, B, 0, 0);
  k_copy2<<<blks((long long)B*512*3), 256, 0, st>>>(l1xyzT, 512*3, u2xyzT, 1024*3, 512*3, B);
  gemm(st, u2T, 128, 512*128, m3Ws, 128, m3bs, 0, u2xyzT + 512*3, 3, 1024*3, 512, 3, 128, B, 0, 0);
  gemm(st, u2T, 128, 512*128, m3W1, 128, m3b1, 0, m3h, 32, 512*32, 512, 32, 128, B, 1, 0);
  gemm(st, m3h, 32,  512*32,  m3W2, 32,  m3b2, 0, u2xyzT + 512*3, 3, 1024*3, 512, 3, 32, B, 0, 1);
  k_copy2<<<blks((long long)B*512*128), 256, 0, st>>>(l1feaT, 512*128, u2feaT, 1024*128, 512*128, B);
  k_copy2<<<blks((long long)B*512*128), 256, 0, st>>>(u2T, 512*128, u2feaT + 512*128, 1024*128, 512*128, B);
  run_transformer(st, u2feaT, u2xyzT, B, 1024, 128, TR4, S);

  // ---- outputs: (l4_fea, u2_xyz, u2_fea) concatenated flat ----
  k_copy<<<blks((long long)B*384), 256, 0, st>>>(out, l4T, (long long)B*384);
  k_nc_to_cn<<<blks((long long)B*1024*3), 256, 0, st>>>(u2xyzT, out + B*384, 1024, 3, (long long)B*1024*3);
  k_nc_to_cn<<<blks((long long)B*1024*128), 256, 0, st>>>(u2feaT, out + B*384 + B*1024*3, 1024, 128, (long long)B*1024*128);
}
